// Seq2Seq_85074712199884
// MI455X (gfx1250) — matrix operand order repaired and checked by reading, not compiled, not run on hardware
//
#include <hip/hip_runtime.h>

// Seq2Seq (3-layer LSTM encoder T=512 + 3-layer LSTM decoder HOR=64) for MI455X
// (gfx1250, wave32). Core: per-cell fused GEMM+LSTM using v_wmma_f32_16x16x32_bf16,
// bf16 operands packed into native B-fragment layout, f32 accumulation.
// B tiles are staged into double-buffered LDS with GLOBAL_LOAD_ASYNC_TO_LDS_B128
// (ASYNCcnt-tracked, gfx1250 async path) so the next tile's fetch overlaps the
// current tile's WMMAs; one barrier + one s_wait_asynccnt per k-tile.

typedef __attribute__((ext_vector_type(16))) __bf16 v16bf;
typedef __attribute__((ext_vector_type(8)))  __bf16 v8bf;
typedef __attribute__((ext_vector_type(8)))  float  v8f;
typedef int v4i __attribute__((vector_size(16)));   // builtin's 16B chunk type

#define T_LEN  512
#define BATCH  256
#define DIN    32
#define HID    512
#define HORZ   64
#define GATES  2048   // 4*HID, torch gate order i,f,g,o

#if defined(__has_builtin)
#if __has_builtin(__builtin_amdgcn_global_load_async_to_lds_b128)
#define ASYNC_STAGE 1
#endif
#endif
#ifndef ASYNC_STAGE
#define ASYNC_STAGE 0
#endif

__device__ __forceinline__ void wait_async_done() {
#if ASYNC_STAGE
#if __has_builtin(__builtin_amdgcn_s_wait_asynccnt)
    __builtin_amdgcn_s_wait_asynccnt(0);
#else
    asm volatile("s_wait_asynccnt 0x0" ::: "memory");
#endif
#endif
}

__device__ __forceinline__ unsigned short f32_to_bf16_rne(float f) {
    unsigned u = __float_as_uint(f);
    unsigned r = u + 0x7FFFu + ((u >> 16) & 1u);   // round-to-nearest-even
    return (unsigned short)(r >> 16);
}
__device__ __forceinline__ float bf16_to_f32(unsigned short h) {
    return __uint_as_float(((unsigned)h) << 16);
}
__device__ __forceinline__ float sigmoidf_(float x) {
    return 1.0f / (1.0f + __expf(-x));
}

// ---- A fragment: 16x32 bf16, rows = batch, from row-major [.,lda] bf16 ----
// lane<16 : M=lane,   K = {0..7, 16..23}
// lane>=16: M=lane-16,K = {8..15,24..31}
__device__ __forceinline__ v16bf load_a_frag(const unsigned short* __restrict__ A,
                                             int lda, int m0, int k0, int lane) {
    int m  = m0 + (lane & 15);
    int kh = lane >> 4;
    const unsigned short* p = A + (size_t)m * lda + k0 + kh * 8;
    v8bf lo = *reinterpret_cast<const v8bf*>(p);
    v8bf hi = *reinterpret_cast<const v8bf*>(p + 16);
    v16bf r;
#pragma unroll
    for (int i = 0; i < 8; ++i) { r[i] = lo[i]; r[i + 8] = hi[i]; }
    return r;
}

// ---- B fragment from LDS-staged packed tile: lane=N (lane&15), K in table order, k-stride 16 ----
__device__ __forceinline__ v16bf load_b_lds(const unsigned short* __restrict__ sB,
                                            int g, int lane) {
    const unsigned short* p = sB + (g << 9) + (lane & 15) + ((lane >> 4) << 7);
    v16bf r;
#pragma unroll
    for (int i = 0; i < 8; ++i) {
        r[i]     = __builtin_bit_cast(__bf16, p[i << 4]);
        r[i + 8] = __builtin_bit_cast(__bf16, p[256 + (i << 4)]);
    }
    return r;
}

// Fused LSTM cell: gates(B x 4H) = A1(B x 32*KT1) @ B1 + A2(B x 32*KT2) @ B2
//                                  [+ xscalar ⊗ wcol] + bias;  then c,h update.
// Grid: 64 blocks x 256 threads (8 waves). Block -> hidden tile ht = blockIdx>>1,
// wave -> M tile mt. Each wave owns a 16x16 block of (batch x hidden) and computes
// its i,f,g,o accumulators (N offsets 0,512,1024,1536) sharing the A fragment.
// The two K streams are concatenated into one pipelined loop so the async-LDS
// double buffer stays primed across the stream boundary.
__global__ __launch_bounds__(256) void lstm_cell_kernel(
    const unsigned short* __restrict__ A1, int lda1,
    const unsigned short* __restrict__ B1, int KT1,
    const unsigned short* __restrict__ A2, int lda2,
    const unsigned short* __restrict__ B2, int KT2,
    const float* __restrict__ xscalar, const float* __restrict__ wcol,
    const float* __restrict__ bias,
    float* __restrict__ c_state, unsigned short* __restrict__ h_out)
{
    __shared__ unsigned short sB[2][2048];       // double-buffered: 2 x (4 gate tiles x 512 bf16)
    const int tid  = threadIdx.x;
    const int lane = tid & 31;
    const int ht   = blockIdx.x >> 1;                              // hidden tile 0..31
    const int mt   = ((blockIdx.x & 1) << 3) + (tid >> 5);         // M tile 0..15

    const int KTtot = KT1 + ((A2 != nullptr) ? KT2 : 0);

    // per-thread staging chunk: gate g = tid>>6, 16B chunk q = tid&63 (4KB per tile-set)
    const int sg = tid >> 6;
    const int sq = tid & 63;

    auto issue_stage = [&](int ktg, int buf) {
        const unsigned short* Bp; int kt, KT;
        if (ktg < KT1) { Bp = B1; kt = ktg;       KT = KT1; }
        else           { Bp = B2; kt = ktg - KT1; KT = KT2; }
        const uint4* src = reinterpret_cast<const uint4*>(
            Bp + ((size_t)((sg * 32 + ht) * KT + kt) << 9)) + sq;
        unsigned short* dst = &sB[buf][tid << 3];
#if ASYNC_STAGE
        __builtin_amdgcn_global_load_async_to_lds_b128(
            (__attribute__((address_space(1))) v4i*)(size_t)src,
            (__attribute__((address_space(3))) v4i*)dst, 0, 0);
#else
        *reinterpret_cast<uint4*>(dst) = *src;
#endif
    };

    v8f acc[4];
#pragma unroll
    for (int g = 0; g < 4; ++g)
#pragma unroll
        for (int r = 0; r < 8; ++r) acc[g][r] = 0.0f;

    issue_stage(0, 0);
    for (int ktg = 0; ktg < KTtot; ++ktg) {
        wait_async_done();      // my chunk of the current tile-set has landed in LDS
        __syncthreads();        // all waves: current buffer complete, previous reads done
        if (ktg + 1 < KTtot) issue_stage(ktg + 1, (ktg + 1) & 1);  // overlap with WMMAs

        v16bf a;
        if (ktg < KT1) a = load_a_frag(A1, lda1, mt << 4, ktg << 5, lane);
        else           a = load_a_frag(A2, lda2, mt << 4, (ktg - KT1) << 5, lane);
        const unsigned short* sBc = &sB[ktg & 1][0];
#pragma unroll
        for (int g = 0; g < 4; ++g) {
            v16bf b = load_b_lds(sBc, g, lane);
            acc[g] = __builtin_amdgcn_wmma_f32_16x16x32_bf16(
                false, a, false, b, (short)0, acc[g], false, false);
        }
    }

    // ---- epilogue: C/D f32 layout -> element (r, lane): M = r + 8*(lane>=16), N = lane&15
    const int n0  = (ht << 4) + (lane & 15);   // hidden index j
    const float bi  = bias[n0];
    const float bfv = bias[512 + n0];
    const float bg  = bias[1024 + n0];
    const float bo  = bias[1536 + n0];
    float wi = 0.f, wf = 0.f, wg = 0.f, wo = 0.f;
    if (wcol) { wi = wcol[n0]; wf = wcol[512 + n0]; wg = wcol[1024 + n0]; wo = wcol[1536 + n0]; }

#pragma unroll
    for (int r = 0; r < 8; ++r) {
        int m = (mt << 4) + ((lane >> 4) << 3) + r;
        float gi = acc[0][r] + bi;
        float gf = acc[1][r] + bfv;
        float gg = acc[2][r] + bg;
        float go = acc[3][r] + bo;
        if (wcol) {                      // rank-1 decoder-L1 input (K=1)
            float xs = xscalar[m];
            gi += xs * wi; gf += xs * wf; gg += xs * wg; go += xs * wo;
        }
        size_t idx = (size_t)m * HID + n0;
        float cn = sigmoidf_(gf) * c_state[idx] + sigmoidf_(gi) * tanhf(gg);
        float hn = sigmoidf_(go) * tanhf(cn);
        c_state[idx] = cn;
        h_out[idx]   = f32_to_bf16_rne(hn);
    }
}

// ---- GEMV head: y[b] = leaky_relu(h[b,:] . w + bias). One wave per batch row. ----
__global__ __launch_bounds__(256) void fc_leaky_kernel(
    const unsigned short* __restrict__ h, const float* __restrict__ w,
    const float* __restrict__ b, float* __restrict__ scal_out,
    float* __restrict__ final_out)
{
    int wv   = (blockIdx.x * 256 + threadIdx.x) >> 5;   // batch row, grid = 32x256 -> 256 waves
    int lane = threadIdx.x & 31;
    const unsigned short* row = h + (size_t)wv * HID;
    float s = 0.f;
#pragma unroll
    for (int k = lane; k < HID; k += 32) s += bf16_to_f32(row[k]) * w[k];
#pragma unroll
    for (int off = 16; off > 0; off >>= 1) s += __shfl_xor(s, off, 32);
    if (lane == 0) {
        float y = s + b[0];
        y = (y > 0.f) ? y : 0.01f * y;
        scal_out[wv] = y;
        if (final_out) final_out[wv] = y;
    }
}

// ---- pack fp32 weight (4H x K row-major) into bf16 B-fragment layout ----
// packed[((nt*KT + kt)*32 + (k&31))*16 + (j&15)] = bf16(W[j,k]); nt=j>>4, kt=k>>5
__global__ __launch_bounds__(256) void pack_weight_kernel(
    const float* __restrict__ in, unsigned short* __restrict__ out, int K)
{
    int idx = blockIdx.x * 256 + threadIdx.x;
    if (idx >= GATES * K) return;
    int j = idx / K, k = idx - j * K;
    int nt = j >> 4, n = j & 15;
    int kt = k >> 5, l = k & 31;
    int KT = K >> 5;
    out[((((nt * KT + kt) << 5) + l) << 4) + n] = f32_to_bf16_rne(in[idx]);
}

__global__ __launch_bounds__(256) void cvt_f32_bf16_kernel(
    const float* __restrict__ in, unsigned short* __restrict__ out, int n)
{
    int i = blockIdx.x * 256 + threadIdx.x;
    if (i < n) out[i] = f32_to_bf16_rne(in[i]);
}

__global__ __launch_bounds__(256) void zero_u32_kernel(unsigned* __restrict__ p, int n)
{
    int i = blockIdx.x * 256 + threadIdx.x;
    if (i < n) p[i] = 0u;
}

extern "C" void kernel_launch(void* const* d_in, const int* in_sizes, int n_in,
                              void* d_out, int out_size, void* d_ws, size_t ws_size,
                              hipStream_t stream)
{
    (void)in_sizes; (void)n_in; (void)out_size; (void)ws_size;
    const float* X    = (const float*)d_in[0];
    // d_in[1] = y : unused by the reference computation
    const float* eW1  = (const float*)d_in[2];
    const float* eU1  = (const float*)d_in[3];
    const float* eb1  = (const float*)d_in[4];
    const float* eW2  = (const float*)d_in[5];
    const float* eU2  = (const float*)d_in[6];
    const float* eb2  = (const float*)d_in[7];
    const float* eW3  = (const float*)d_in[8];
    const float* eU3  = (const float*)d_in[9];
    const float* eb3  = (const float*)d_in[10];
    const float* dW1  = (const float*)d_in[11];
    const float* dU1  = (const float*)d_in[12];
    const float* db1  = (const float*)d_in[13];
    const float* dW2  = (const float*)d_in[14];
    const float* dU2  = (const float*)d_in[15];
    const float* db2  = (const float*)d_in[16];
    const float* dW3  = (const float*)d_in[17];
    const float* dU3  = (const float*)d_in[18];
    const float* db3  = (const float*)d_in[19];
    const float* fceW = (const float*)d_in[20];
    const float* fceB = (const float*)d_in[21];
    const float* fcdW = (const float*)d_in[22];
    const float* fcdB = (const float*)d_in[23];
    float* out = (float*)d_out;

    // ---- workspace bump allocator (256B aligned); total ~35 MB ----
    char* ws = (char*)d_ws;
    size_t off = 0;
    auto alloc = [&](size_t bytes) -> char* {
        char* p = ws + off;
        off = (off + bytes + 255) & ~(size_t)255;
        return p;
    };

    unsigned short* Xbf  = (unsigned short*)alloc((size_t)T_LEN * BATCH * DIN * 2);
    unsigned short* eW1p = (unsigned short*)alloc((size_t)GATES * DIN * 2);
    unsigned short* eU1p = (unsigned short*)alloc((size_t)GATES * HID * 2);
    unsigned short* eW2p = (unsigned short*)alloc((size_t)GATES * HID * 2);
    unsigned short* eU2p = (unsigned short*)alloc((size_t)GATES * HID * 2);
    unsigned short* eW3p = (unsigned short*)alloc((size_t)GATES * HID * 2);
    unsigned short* eU3p = (unsigned short*)alloc((size_t)GATES * HID * 2);
    unsigned short* dU1p = (unsigned short*)alloc((size_t)GATES * HID * 2);
    unsigned short* dW2p = (unsigned short*)alloc((size_t)GATES * HID * 2);
    unsigned short* dU2p = (unsigned short*)alloc((size_t)GATES * HID * 2);
    unsigned short* dW3p = (unsigned short*)alloc((size_t)GATES * HID * 2);
    unsigned short* dU3p = (unsigned short*)alloc((size_t)GATES * HID * 2);

    char* state_base = ws + off;
    unsigned short* hbuf[3][2];
    for (int l = 0; l < 3; ++l)
        for (int p = 0; p < 2; ++p)
            hbuf[l][p] = (unsigned short*)alloc((size_t)BATCH * HID * 2);
    float* cbuf[3];
    for (int l = 0; l < 3; ++l) cbuf[l] = (float*)alloc((size_t)BATCH * HID * 4);
    size_t state_bytes = (size_t)((ws + off) - state_base);
    float* sbuf = (float*)alloc((size_t)BATCH * 4);   // decoder scalar input

    // ---- prep: zero states, convert X, pack weights (L2-resident thereafter) ----
    {
        int n32 = (int)(state_bytes >> 2);
        zero_u32_kernel<<<(n32 + 255) / 256, 256, 0, stream>>>((unsigned*)state_base, n32);
        int nx = T_LEN * BATCH * DIN;
        cvt_f32_bf16_kernel<<<(nx + 255) / 256, 256, 0, stream>>>(X, Xbf, nx);
        auto packW = [&](const float* w, unsigned short* p, int K) {
            int total = GATES * K;
            pack_weight_kernel<<<(total + 255) / 256, 256, 0, stream>>>(w, p, K);
        };
        packW(eW1, eW1p, DIN);
        packW(eU1, eU1p, HID); packW(eW2, eW2p, HID); packW(eU2, eU2p, HID);
        packW(eW3, eW3p, HID); packW(eU3, eU3p, HID);
        packW(dU1, dU1p, HID); packW(dW2, dW2p, HID); packW(dU2, dU2p, HID);
        packW(dW3, dW3p, HID); packW(dU3, dU3p, HID);
    }

    const dim3 cg(64), cb(256);
    int par = 0;

    // ---- encoder: 512 steps x 3 stacked cells (serial dependence on stream) ----
    for (int t = 0; t < T_LEN; ++t) {
        const unsigned short* xt = Xbf + (size_t)t * BATCH * DIN;
        lstm_cell_kernel<<<cg, cb, 0, stream>>>(
            hbuf[0][par], HID, eU1p, HID / 32, xt, DIN, eW1p, DIN / 32,
            nullptr, nullptr, eb1, cbuf[0], hbuf[0][par ^ 1]);
        lstm_cell_kernel<<<cg, cb, 0, stream>>>(
            hbuf[1][par], HID, eU2p, HID / 32, hbuf[0][par ^ 1], HID, eW2p, HID / 32,
            nullptr, nullptr, eb2, cbuf[1], hbuf[1][par ^ 1]);
        lstm_cell_kernel<<<cg, cb, 0, stream>>>(
            hbuf[2][par], HID, eU3p, HID / 32, hbuf[1][par ^ 1], HID, eW3p, HID / 32,
            nullptr, nullptr, eb3, cbuf[2], hbuf[2][par ^ 1]);
        par ^= 1;
    }

    // ---- seq2seq head on last encoder output -> decoder seed ----
    fc_leaky_kernel<<<32, 256, 0, stream>>>(hbuf[2][par], fceW, fceB, sbuf, nullptr);

    // ---- decoder: 64 autoregressive steps (states continue from encoder) ----
    for (int s = 0; s < HORZ; ++s) {
        lstm_cell_kernel<<<cg, cb, 0, stream>>>(
            hbuf[0][par], HID, dU1p, HID / 32, nullptr, 0, nullptr, 0,
            sbuf, dW1, db1, cbuf[0], hbuf[0][par ^ 1]);
        lstm_cell_kernel<<<cg, cb, 0, stream>>>(
            hbuf[1][par], HID, dU2p, HID / 32, hbuf[0][par ^ 1], HID, dW2p, HID / 32,
            nullptr, nullptr, db2, cbuf[1], hbuf[1][par ^ 1]);
        lstm_cell_kernel<<<cg, cb, 0, stream>>>(
            hbuf[2][par], HID, dU3p, HID / 32, hbuf[1][par ^ 1], HID, dW3p, HID / 32,
            nullptr, nullptr, db3, cbuf[2], hbuf[2][par ^ 1]);
        fc_leaky_kernel<<<32, 256, 0, stream>>>(
            hbuf[2][par ^ 1], fcdW, fcdB, sbuf, out + (size_t)s * BATCH);
        par ^= 1;
    }
}
